// GCN_4277787427615
// MI455X (gfx1250) — matrix operand order repaired and checked by reading, not compiled, not run on hardware
//
#include <hip/hip_runtime.h>

typedef __attribute__((ext_vector_type(16))) _Float16 v16h;
typedef __attribute__((ext_vector_type(8)))  _Float16 v8h;
typedef __attribute__((ext_vector_type(8)))  float    v8f;

#define NNODES 100000
#define DH 128

// ---------------- utility kernels ----------------

__global__ void k_zero_f32(float* __restrict__ p, long long n) {
  long long i = (long long)blockIdx.x * blockDim.x + threadIdx.x;
  if (i < n) p[i] = 0.0f;
}

__global__ void k_init_deg(float* __restrict__ deg, int n) {
  int i = blockIdx.x * blockDim.x + threadIdx.x;
  if (i < n) deg[i] = 1.0f;  // self loop contributes 1 to every node's degree
}

__global__ void k_count_deg(const int* __restrict__ dst, long long E, float* __restrict__ deg) {
  long long e = (long long)blockIdx.x * blockDim.x + threadIdx.x;
  if (e < E) atomicAdd(&deg[dst[e]], 1.0f);
}

__global__ void k_rsqrt_inplace(float* __restrict__ d, int n) {
  int i = blockIdx.x * blockDim.x + threadIdx.x;
  if (i < n) d[i] = rsqrtf(d[i]);
}

__global__ void k_f32_to_f16(const float* __restrict__ a, _Float16* __restrict__ b, long long n) {
  long long i = (long long)blockIdx.x * blockDim.x + threadIdx.x;
  if (i < n) b[i] = (_Float16)a[i];
}

// Wt[n][k] = W[k][n], zero-padded to ncols_pad rows. K fixed = 128.
__global__ void k_transpose_w(const float* __restrict__ W, _Float16* __restrict__ Wt,
                              int ncols_real, int ncols_pad) {
  int idx = blockIdx.x * blockDim.x + threadIdx.x;
  if (idx >= ncols_pad * DH) return;
  int n = idx >> 7;        // row of Wt (output column)
  int k = idx & (DH - 1);  // k index
  Wt[idx] = (n < ncols_real) ? (_Float16)W[(size_t)k * ncols_real + n] : (_Float16)0.0f;
}

// ---------------- WMMA GEMM: H[N x ncols] = A[N x 128] * W[128 x ncols] ----------------
// A: f16 row-major [N][128].  Wt: f16, transposed weights [ncols_pad][128].
// One wave computes RT row-tiles x one 16-wide col tile; B fragment reused across row tiles.
template <int RT>
__global__ __launch_bounds__(32) void k_gemm_wmma(const _Float16* __restrict__ A,
                                                  const _Float16* __restrict__ Wt,
                                                  float* __restrict__ H,
                                                  int ldh, int ncols) {
  const int lane = threadIdx.x & 31;
  const int half = lane >> 4;   // 0: lanes 0-15, 1: lanes 16-31
  const int ml   = lane & 15;
  const int row0 = blockIdx.x * (16 * RT);
  const int col0 = blockIdx.y * 16;

  const _Float16* __restrict__ abase = A + (size_t)(row0 + ml) * DH;          // M = lane%16
  const _Float16* __restrict__ brow  = Wt + (size_t)(col0 + ml) * DH;         // N = lane%16

  v8f acc[RT];
#pragma unroll
  for (int t = 0; t < RT; ++t) acc[t] = (v8f){};

#pragma unroll
  for (int kc = 0; kc < 4; ++kc) {
    const int k0 = kc * 32;
    // B 32x16 f16 fragment: lanes 0-15 hold K = 0..15, lanes 16-31 hold K = 16..31
    v16h b = *(const v16h*)(brow + k0 + half * 16);
#pragma unroll
    for (int t = 0; t < RT; ++t) {
      const _Float16* __restrict__ arow = abase + (size_t)t * 16 * DH;
      // A 16x32 f16 fragment: VGPR0-3 hold K = 16*half + 0..7 ; VGPR4-7 hold K = 16*half + 8..15
      v8h alo = *(const v8h*)(arow + k0 + half * 16);
      v8h ahi = *(const v8h*)(arow + k0 + half * 16 + 8);
      v16h a;
#pragma unroll
      for (int i = 0; i < 8; ++i) { a[i] = alo[i]; a[8 + i] = ahi[i]; }
      acc[t] = __builtin_amdgcn_wmma_f32_16x16x32_f16(false, a, false, b, (short)0, acc[t],
                                                      false, false);
    }
  }

  const int col = col0 + ml;
  if (col < ncols) {
#pragma unroll
    for (int t = 0; t < RT; ++t) {
#pragma unroll
      for (int r = 0; r < 8; ++r) {
        // C/D layout: VGPR r holds M = r (lanes 0-15) / M = r+8 (lanes 16-31), N = lane%16
        H[(size_t)(row0 + t * 16 + r + half * 8) * ldh + col] = acc[t][r];
      }
    }
  }
}

// ------- edge scatter, d = 128, vectorized: agg[dst] += h[src] * dinv[src]*dinv[dst] -------
// warp per edge; lane l owns features 4l..4l+3 (one b128 load, 4 contiguous f32 atomics)
__global__ void k_scatter128(const int* __restrict__ src, const int* __restrict__ dst,
                             const float* __restrict__ h, const float* __restrict__ dinv,
                             float* __restrict__ agg, long long E) {
  long long gid = (long long)blockIdx.x * blockDim.x + threadIdx.x;
  long long e = gid >> 5;
  int lane = (int)(gid & 31);
  if (e >= E) return;
  int s = src[e], t = dst[e];
  float w = dinv[s] * dinv[t];
  const float4 v = *(const float4*)(h + (size_t)s * DH + lane * 4);
  float* __restrict__ at = agg + (size_t)t * DH + lane * 4;
  atomicAdd(at + 0, v.x * w);
  atomicAdd(at + 1, v.y * w);
  atomicAdd(at + 2, v.z * w);
  atomicAdd(at + 3, v.w * w);
}

// ------- generic edge scatter (used for d = 40) -------
__global__ void k_scatter(const int* __restrict__ src, const int* __restrict__ dst,
                          const float* __restrict__ h, const float* __restrict__ dinv,
                          float* __restrict__ agg, int d, long long E) {
  long long gid = (long long)blockIdx.x * blockDim.x + threadIdx.x;
  long long e = gid >> 5;
  int lane = (int)(gid & 31);
  if (e >= E) return;
  int s = src[e], t = dst[e];
  float w = dinv[s] * dinv[t];
  const float* __restrict__ hs = h + (size_t)s * d;
  float* __restrict__ at = agg + (size_t)t * d;
  for (int f = lane; f < d; f += 32)
    atomicAdd(&at[f], hs[f] * w);
}

// ---------------- self loop + bias + BN stats (d = 128) ----------------
// v = agg + h*dinv^2 + b ; agg <- v ; stats[f] += sum v, stats[128+f] += sum v^2
__global__ __launch_bounds__(128) void k_finalize_stats(float* __restrict__ agg,
                                                        const float* __restrict__ h,
                                                        const float* __restrict__ dinv,
                                                        const float* __restrict__ b,
                                                        float* __restrict__ stats,
                                                        int N, int rows_per_block) {
  const int f = threadIdx.x;  // 0..127
  const float bf = b[f];
  int r0 = blockIdx.x * rows_per_block;
  int r1 = r0 + rows_per_block; if (r1 > N) r1 = N;
  float s1 = 0.0f, s2 = 0.0f;
  for (int r = r0; r < r1; ++r) {
    float di = dinv[r];
    size_t idx = (size_t)r * DH + f;
    float v = agg[idx] + h[idx] * di * di + bf;
    agg[idx] = v;
    s1 += v;
    s2 += v * v;
  }
  atomicAdd(&stats[f], s1);
  atomicAdd(&stats[DH + f], s2);
}

// ---------------- BN apply + ReLU, write f16 for next GEMM ----------------
__global__ void k_bn_apply(const float* __restrict__ agg, const float* __restrict__ stats,
                           const float* __restrict__ g, const float* __restrict__ be,
                           _Float16* __restrict__ yh, int N) {
  long long idx = (long long)blockIdx.x * blockDim.x + threadIdx.x;
  if (idx >= (long long)N * DH) return;
  int f = (int)(idx & (DH - 1));
  const float invN = 1.0f / (float)N;
  float mu  = stats[f] * invN;
  float var = stats[DH + f] * invN - mu * mu;
  float v = (agg[idx] - mu) * rsqrtf(var + 1e-5f) * g[f] + be[f];
  yh[idx] = (_Float16)fmaxf(v, 0.0f);
}

// ---------------- layer3 finalize + log_softmax (d = 40), warp per row ----------------
__global__ __launch_bounds__(256) void k_finalize_lsm(const float* __restrict__ agg,
                                                      const float* __restrict__ h,
                                                      const float* __restrict__ dinv,
                                                      const float* __restrict__ b,
                                                      float* __restrict__ out, int N) {
  const int row = blockIdx.x * (blockDim.x >> 5) + (threadIdx.x >> 5);
  const int lane = threadIdx.x & 31;
  if (row >= N) return;
  float di = dinv[row];
  float sl = di * di;
  size_t base = (size_t)row * 40;
  float v0 = agg[base + lane] + h[base + lane] * sl + b[lane];
  float v1 = -3.0e38f;
  if (lane < 8) v1 = agg[base + 32 + lane] + h[base + 32 + lane] * sl + b[32 + lane];
  float m = fmaxf(v0, v1);
#pragma unroll
  for (int off = 16; off > 0; off >>= 1) m = fmaxf(m, __shfl_xor(m, off, 32));
  float s = __expf(v0 - m) + ((lane < 8) ? __expf(v1 - m) : 0.0f);
#pragma unroll
  for (int off = 16; off > 0; off >>= 1) s += __shfl_xor(s, off, 32);
  float lse = m + __logf(s);
  out[base + lane] = v0 - lse;
  if (lane < 8) out[base + 32 + lane] = v1 - lse;
}

// ---------------- host orchestration ----------------

extern "C" void kernel_launch(void* const* d_in, const int* in_sizes, int n_in,
                              void* d_out, int out_size, void* d_ws, size_t ws_size,
                              hipStream_t stream) {
  (void)n_in; (void)out_size; (void)ws_size;
  const float* x   = (const float*)d_in[0];
  const float* W1  = (const float*)d_in[1];
  const float* b1  = (const float*)d_in[2];
  const float* W2  = (const float*)d_in[3];
  const float* b2  = (const float*)d_in[4];
  const float* W3  = (const float*)d_in[5];
  const float* b3  = (const float*)d_in[6];
  const float* g1  = (const float*)d_in[7];
  const float* be1 = (const float*)d_in[8];
  const float* g2  = (const float*)d_in[9];
  const float* be2 = (const float*)d_in[10];
  const int*   ei  = (const int*)d_in[11];   // [2, E] per harness int staging
  const long long E = (long long)in_sizes[11] / 2;
  const int* srcI = ei;
  const int* dstI = ei + E;
  const int N = NNODES;
  float* out = (float*)d_out;

  // carve workspace
  char* wsp = (char*)d_ws;
  size_t off = 0;
  auto carve = [&](size_t bytes) -> char* {
    char* p = wsp + off;
    off = (off + bytes + 255) & ~(size_t)255;
    return p;
  };
  float*    dinv  = (float*)   carve((size_t)N * 4);
  _Float16* Xh    = (_Float16*)carve((size_t)N * DH * 2);   // reused as yh between layers
  float*    h     = (float*)   carve((size_t)N * DH * 4);   // GEMM output (layer3 uses [N][40] subset)
  float*    agg   = (float*)   carve((size_t)N * DH * 4);
  _Float16* Wt1   = (_Float16*)carve(128 * 128 * 2);
  _Float16* Wt2   = (_Float16*)carve(128 * 128 * 2);
  _Float16* Wt3   = (_Float16*)carve(48 * 128 * 2);
  float*    stats = (float*)   carve(2 * DH * 4);

  const int T = 256;
  const long long ND = (long long)N * DH;

  // degrees / dinv
  k_init_deg<<<(N + T - 1) / T, T, 0, stream>>>(dinv, N);
  k_count_deg<<<(int)((E + T - 1) / T), T, 0, stream>>>(dstI, E, dinv);
  k_rsqrt_inplace<<<(N + T - 1) / T, T, 0, stream>>>(dinv, N);

  // precompute f16 operands
  k_f32_to_f16<<<(int)((ND + T - 1) / T), T, 0, stream>>>(x, Xh, ND);
  k_transpose_w<<<(128 * 128 + T - 1) / T, T, 0, stream>>>(W1, Wt1, 128, 128);
  k_transpose_w<<<(128 * 128 + T - 1) / T, T, 0, stream>>>(W2, Wt2, 128, 128);
  k_transpose_w<<<(48 * 128 + T - 1) / T, T, 0, stream>>>(W3, Wt3, 40, 48);

  // 6250 row tiles = 5 * 1250 exactly -> 5 row tiles per wave
  dim3 gemm_grid12(N / (16 * 5), 8);   // 128 output cols
  dim3 gemm_grid3(N / (16 * 5), 3);    // 48 padded cols (40 real)
  const long long sthreads = E * 32;
  const int sblocks = (int)((sthreads + T - 1) / T);

  // ---- layer 1 ----
  k_gemm_wmma<5><<<gemm_grid12, 32, 0, stream>>>(Xh, Wt1, h, 128, 128);
  k_zero_f32<<<(int)((ND + T - 1) / T), T, 0, stream>>>(agg, ND);
  k_zero_f32<<<1, 256, 0, stream>>>(stats, 2 * DH);
  k_scatter128<<<sblocks, T, 0, stream>>>(srcI, dstI, h, dinv, agg, E);
  k_finalize_stats<<<(N + 255) / 256, 128, 0, stream>>>(agg, h, dinv, b1, stats, N, 256);
  k_bn_apply<<<(int)((ND + T - 1) / T), T, 0, stream>>>(agg, stats, g1, be1, Xh, N);

  // ---- layer 2 ----
  k_gemm_wmma<5><<<gemm_grid12, 32, 0, stream>>>(Xh, Wt2, h, 128, 128);
  k_zero_f32<<<(int)((ND + T - 1) / T), T, 0, stream>>>(agg, ND);
  k_zero_f32<<<1, 256, 0, stream>>>(stats, 2 * DH);
  k_scatter128<<<sblocks, T, 0, stream>>>(srcI, dstI, h, dinv, agg, E);
  k_finalize_stats<<<(N + 255) / 256, 128, 0, stream>>>(agg, h, dinv, b2, stats, N, 256);
  k_bn_apply<<<(int)((ND + T - 1) / T), T, 0, stream>>>(agg, stats, g2, be2, Xh, N);

  // ---- layer 3 ----
  k_gemm_wmma<5><<<gemm_grid3, 32, 0, stream>>>(Xh, Wt3, h, 40, 40);
  const long long n40 = (long long)N * 40;
  k_zero_f32<<<(int)((n40 + T - 1) / T), T, 0, stream>>>(agg, n40);
  k_scatter<<<sblocks, T, 0, stream>>>(srcI, dstI, h, dinv, agg, 40, E);
  k_finalize_lsm<<<(N + 7) / 8, 256, 0, stream>>>(agg, h, dinv, b3, out, N);
}
